// FrameFrechetMeanCosineDistanceNetwork_75565654606141
// MI455X (gfx1250) — compile-verified, matrix operand order checked by reading, not run on hardware
//
#include <hip/hip_runtime.h>
#include <math.h>

// Problem constants (from reference)
#define DF       2048      // per-frame dim
#define NF       8         // frames
#define NITERS   32
#define NTHREADS 256
#define NWAVES   8         // 256 / wave32
#define DPT      8         // DF / NTHREADS, dims owned per thread
#define CPW      8         // K=32 chunks per wave: (DF/NWAVES)/32

typedef __attribute__((ext_vector_type(16))) _Float16 v16h;
typedef __attribute__((ext_vector_type(8)))  _Float16 v8h;
typedef __attribute__((ext_vector_type(8)))  float    v8f;
typedef __attribute__((ext_vector_type(4)))  float    v4f;

struct Smem {
  union {
    float stage[NF * DF];          // 64 KB f32 async-staging buffer
    struct {
      _Float16 X[NF][DF];          // 32 KB normalized frames (f16)
      _Float16 mu16[DF];           // 4 KB current mean (f16)
    } h;
  } u;
  float red[NWAVES];
  float dotsPart[NWAVES][NF];
};

__device__ __forceinline__ float block_reduce_sum(float v, float* red) {
  #pragma unroll
  for (int off = 16; off > 0; off >>= 1) v += __shfl_xor(v, off, 32);
  const int w = threadIdx.x >> 5;
  __syncthreads();                       // protect red reuse
  if ((threadIdx.x & 31) == 0) red[w] = v;
  __syncthreads();
  float t = 0.f;
  #pragma unroll
  for (int i = 0; i < NWAVES; ++i) t += red[i];
  return t;
}

// Stage NF*DF f32 (64 KB) from global into LDS via the async-DMA path
// (global_load_async_to_lds_b128, tracked on ASYNCcnt — no VGPR round trip).
__device__ __forceinline__ void async_stage(const float* __restrict__ src,
                                            float* stage) {
  const uint32_t lds_base = (uint32_t)(uintptr_t)stage;  // low 32 bits = LDS offset
  const uint64_t gbase    = (uint64_t)(uintptr_t)src;
  const int t = threadIdx.x;
  #pragma unroll
  for (int i = 0; i < (NF * DF * 4) / (NTHREADS * 16); ++i) {   // 16 x 16B per thread
    const uint32_t off = (uint32_t)((i * NTHREADS + t) * 16);
    const uint32_t l = lds_base + off;
    const uint64_t g = gbase + off;
    asm volatile("global_load_async_to_lds_b128 %0, %1, off"
                 :: "v"(l), "v"(g) : "memory");
  }
  asm volatile("s_wait_asynccnt 0x0" ::: "memory");
  __syncthreads();
}

__device__ __forceinline__ v16h frag_gather(const _Float16* p) {
  // 8 contiguous halves at p (16B aligned) and 8 halves at p+16 -> two b128 loads
  const v8h lo = *(const v8h*)p;
  const v8h hi = *(const v8h*)(p + 16);
  return __builtin_shufflevector(lo, hi, 0, 1, 2, 3, 4, 5, 6, 7,
                                         8, 9, 10, 11, 12, 13, 14, 15);
}

// Fréchet mean of NF rows of src ([NF][DF], row-major) on the unit sphere.
// On exit: thread t holds mu[d] for d = t*DPT .. t*DPT+7 (f32).
__device__ void frechet_mean(const float* __restrict__ src, Smem& s, float mu[DPT]) {
  const int tid  = threadIdx.x;
  const int d0   = tid * DPT;
  const int wave = tid >> 5;
  const int lane = tid & 31;

  // ---- Phase 0: async DMA the raw f32 frames into LDS
  async_stage(src, s.u.stage);

  // ---- Phase 1: normalize each frame -> LDS f16; accumulate mean for mu0.
  // Overwrite order vs. the staging union is barrier-safe (see analysis).
  float macc[DPT];
  #pragma unroll
  for (int i = 0; i < DPT; ++i) macc[i] = 0.f;
  for (int k = 0; k < NF; ++k) {
    const v4f x0 = *(const v4f*)&s.u.stage[k * DF + d0];
    const v4f x1 = *(const v4f*)&s.u.stage[k * DF + d0 + 4];
    float xv[DPT], p = 0.f;
    #pragma unroll
    for (int i = 0; i < 4; ++i) { xv[i] = x0[i]; xv[4 + i] = x1[i]; }
    #pragma unroll
    for (int i = 0; i < DPT; ++i) p += xv[i] * xv[i];
    const float n2  = block_reduce_sum(p, s.red);   // all stage reads of frame k done here
    const float inv = 1.f / fmaxf(sqrtf(n2), 1e-7f);
    v8h xs;
    #pragma unroll
    for (int i = 0; i < DPT; ++i) {
      const float xn = xv[i] * inv;
      macc[i] += xn;
      xs[i] = (_Float16)xn;
    }
    *(v8h*)&s.u.h.X[k][d0] = xs;
  }
  __syncthreads();                       // X fully written before fragment build

  // ---- Phase 2: per-wave WMMA A-fragments (loop-invariant: built once)
  // gfx1250 f16 A 16x32 layout: lane L holds row M = L%16,
  //   element j -> K = (j&7) + (j>=8 ? 16 : 0) + (L>=16 ? 8 : 0)
  const int row  = lane & 15;
  const int hsel = (lane >= 16) ? 8 : 0;
  const int dw   = wave * (CPW * 32);    // this wave's 256-dim slice
  v16h afrag[CPW];
  const v16h zfrag = {};
  #pragma unroll
  for (int c = 0; c < CPW; ++c) {
    const int db = dw + c * 32;
    afrag[c] = (row < NF) ? frag_gather(&s.u.h.X[row][db + hsel]) : zfrag;
  }

  // ---- mu0 = normalize(mean over frames)
  {
    float p = 0.f;
    #pragma unroll
    for (int i = 0; i < DPT; ++i) { macc[i] *= (1.f / NF); p += macc[i] * macc[i]; }
    const float n2  = block_reduce_sum(p, s.red);
    const float inv = 1.f / fmaxf(sqrtf(n2), 1e-7f);
    v8h ms;
    #pragma unroll
    for (int i = 0; i < DPT; ++i) { mu[i] = macc[i] * inv; ms[i] = (_Float16)mu[i]; }
    *(v8h*)&s.u.h.mu16[d0] = ms;
  }
  __syncthreads();

  // ---- Phase 3: fixed-point iterations mu <- Exp_mu(mean_k Log_mu(x_k))
  for (int it = 0; it < NITERS; ++it) {
    // dots[k] = <mu, X_k> on the matrix pipe. B = mu chunk replicated across
    // all 16 N columns, so D[r, n] == dots[r] for every column n; frames on M.
    v8f acc = {};
    #pragma unroll
    for (int c = 0; c < CPW; ++c) {
      const v16h b = frag_gather(&s.u.h.mu16[dw + c * 32 + hsel]);
      acc = __builtin_amdgcn_wmma_f32_16x16x32_f16(false, afrag[c], false, b,
                                                   (short)0, acc, false, false);
    }
    if (lane == 0) {                     // lane 0 holds D[0..7, col0] = dots[0..7]
      #pragma unroll
      for (int r = 0; r < NF; ++r) s.dotsPart[wave][r] = acc[r];
    }
    __syncthreads();

    float coef[NF], s1 = 0.f;
    #pragma unroll
    for (int k = 0; k < NF; ++k) {
      float dsum = 0.f;
      #pragma unroll
      for (int w = 0; w < NWAVES; ++w) dsum += s.dotsPart[w][k];
      dsum = fminf(fmaxf(dsum, -1.f + 1e-6f), 1.f - 1e-6f);
      const float th = acosf(dsum);
      coef[k] = (th > 1e-4f) ? (th / sinf(th)) : 1.f;
      s1 += coef[k] * dsum;
    }

    // a[d] = sum_k coef_k X_k[d]  (one b128 LDS load per frame)
    float a[DPT];
    #pragma unroll
    for (int i = 0; i < DPT; ++i) a[i] = 0.f;
    #pragma unroll
    for (int k = 0; k < NF; ++k) {
      const v8h xk = *(const v8h*)&s.u.h.X[k][d0];
      #pragma unroll
      for (int i = 0; i < DPT; ++i) a[i] += coef[k] * (float)xk[i];
    }
    // v[d] = (1/NF) * (a[d] - s1 * mu[d]);  exp map + renormalize
    float v[DPT], p = 0.f;
    #pragma unroll
    for (int i = 0; i < DPT; ++i) {
      v[i] = (a[i] - s1 * mu[i]) * (1.f / NF);
      p += v[i] * v[i];
    }
    const float vn   = sqrtf(block_reduce_sum(p, s.red));
    const float sinc = (vn > 1e-9f) ? (sinf(vn) / vn) : 1.f;
    const float cv   = cosf(vn);
    float q = 0.f;
    #pragma unroll
    for (int i = 0; i < DPT; ++i) { mu[i] = cv * mu[i] + sinc * v[i]; q += mu[i] * mu[i]; }
    const float inv = 1.f / fmaxf(sqrtf(block_reduce_sum(q, s.red)), 1e-7f);
    v8h ms;
    #pragma unroll
    for (int i = 0; i < DPT; ++i) { mu[i] *= inv; ms[i] = (_Float16)mu[i]; }
    *(v8h*)&s.u.h.mu16[d0] = ms;
    __syncthreads();
  }
}

__global__ void __launch_bounds__(NTHREADS)
target_mean_kernel(const float* __restrict__ tgt, float* __restrict__ tmeans) {
  __shared__ Smem s;
  const int b = blockIdx.x;
  float mu[DPT];
  frechet_mean(tgt + (size_t)b * (NF * DF), s, mu);
  const int d0 = threadIdx.x * DPT;
  #pragma unroll
  for (int i = 0; i < DPT; ++i) tmeans[(size_t)b * DF + d0 + i] = mu[i];
}

__global__ void __launch_bounds__(NTHREADS)
support_sim_kernel(const float* __restrict__ sup, const float* __restrict__ tmeans,
                   float* __restrict__ out, int bs) {
  __shared__ Smem s;
  const int sb = blockIdx.x;            // = s*bs + b
  const int b  = sb % bs;
  // warm the target-mean row while we crunch (global_prefetch_b8)
  __builtin_prefetch(tmeans + (size_t)b * DF + threadIdx.x * DPT, 0, 1);
  float mu[DPT];
  frechet_mean(sup + (size_t)sb * (NF * DF), s, mu);
  const int d0 = threadIdx.x * DPT;
  float p = 0.f;
  #pragma unroll
  for (int i = 0; i < DPT; ++i) p += mu[i] * tmeans[(size_t)b * DF + d0 + i];
  const float sim = block_reduce_sum(p, s.red);
  if (threadIdx.x == 0) out[sb] = sim;
}

extern "C" void kernel_launch(void* const* d_in, const int* in_sizes, int n_in,
                              void* d_out, int out_size, void* d_ws, size_t ws_size,
                              hipStream_t stream) {
  const float* sup = (const float*)d_in[0];   // [25, 64, 8*2048]
  const float* tgt = (const float*)d_in[1];   // [64, 8*2048]
  float* out    = (float*)d_out;              // [25, 64]
  float* tmeans = (float*)d_ws;               // [64, 2048] scratch (512 KB)

  const int per_batch = NF * DF;              // 16384
  const int bs = in_sizes[1] / per_batch;     // 64
  const int nb = in_sizes[0] / per_batch;     // 1600 = 25*64

  target_mean_kernel<<<bs, NTHREADS, 0, stream>>>(tgt, tmeans);
  support_sim_kernel<<<nb, NTHREADS, 0, stream>>>(sup, tmeans, out, bs);
}
